// HiPPO_89103391523689
// MI455X (gfx1250) — compile-verified, matrix operand order checked by reading, not run on hardware
//
#include <hip/hip_runtime.h>

// ---------------------------------------------------------------------------
// HiPPO-LegT linear recurrence  c[t] = c[t-1] @ dA^T + inputs[t] * dB
// L=2048, B=256, N=256, out (L,B,N) f32 = 512 MiB.
//
// MI455X (gfx1250, wave32) strategy:
//  * two-pass chunked scan: 16 chunks x 128 steps -> 4096 concurrent waves
//  * per step: (16b x 256k) @ (256k x 256n) via v_wmma_f32_16x16x32_bf16,
//    hi/lo bf16 split of both dA and the state (3 products ~ fp32 precision)
//  * 3 independent WMMA accumulator chains (8 deep each) for matrix-pipe ILP
//  * ping-pong LDS state buffers -> 1 barrier per step
//  * dA^128 for the chunk combine computed on-device (7 repeated squarings)
//  * step-invariant B-operands (dA^T tiles) pinned in 128 VGPRs per wave
// ---------------------------------------------------------------------------

typedef __attribute__((ext_vector_type(16))) __bf16 v16bf;
typedef __attribute__((ext_vector_type(8)))  __bf16 v8bf;
typedef __attribute__((ext_vector_type(8)))  float  v8f;

#define LSEQ   2048
#define BATCH  256
#define NORD   256
#define NCHUNK 16
#define LCHUNK 128

// ---------------- small fp32 matmul for repeated squaring of dA -------------
__global__ void hippo_sq(const float* A, const float* Bm, float* C) {
  int j = blockIdx.x * 16 + threadIdx.x;
  int i = blockIdx.y * 16 + threadIdx.y;
  float s = 0.f;
  for (int k = 0; k < NORD; ++k) s += A[i * NORD + k] * Bm[k * NORD + j];
  C[i * NORD + j] = s;
}

// -------- transpose + bf16 hi/lo split: out[k][n] = src[n][k] ---------------
__global__ void hippo_tsplit(const float* __restrict__ src,
                             __bf16* __restrict__ hi, __bf16* __restrict__ lo) {
  int idx = blockIdx.x * 256 + threadIdx.x;  // enumerates (k, n)
  int k = idx >> 8;
  int n = idx & 255;
  float x = src[n * NORD + k];
  __bf16 h = (__bf16)x;
  hi[k * NORD + n] = h;
  lo[k * NORD + n] = (__bf16)(x - (float)h);
}

// A-fragment (16x32 bf16) from LDS, row-major state layout.
// lane 0-15: row M=lane,   K = {0..7, 16..23} (+ktile*32)
// lane16-31: row M=lane-16,K = {8..15,24..31} (+ktile*32)
__device__ __forceinline__ v16bf load_a_frag(const __bf16* p0) {
  v8bf a0 = *(const v8bf*)(p0);        // ds_load_b128
  v8bf a1 = *(const v8bf*)(p0 + 16);   // ds_load_b128
  v16bf a;
#pragma unroll
  for (int j = 0; j < 8; ++j) { a[j] = a0[j]; a[j + 8] = a1[j]; }
  return a;
}

// ---------------------------------------------------------------------------
// MODE 1: rank-1 u (inputs*dB), zero init,   store chunk-final state only
// MODE 2: full  u (= chunk finals), zero init, store all states (chunk combine)
// MODE 3: rank-1 u, init from carry states,  store all steps to d_out
// ---------------------------------------------------------------------------
template <int MODE, int STEPS>
__global__ __launch_bounds__(512, 1) void hippo_scan(
    const float* __restrict__ uSrc,   // inputs (MODE1/3) or chunk finals (MODE2)
    const float* __restrict__ dB,     // MODE1/3
    const __bf16* __restrict__ Bh,    // (dA or dA^128)^T, bf16 hi, [k][n]
    const __bf16* __restrict__ Bl,    // bf16 lo
    const float* __restrict__ initS,  // MODE3: carry states (NCHUNK,B,N)
    float* __restrict__ outAll,       // MODE2/3
    float* __restrict__ outFinal) {   // MODE1
  // Ping-pong state buffers: read buf[t&1], write buf[(t+1)&1].
  __shared__ __attribute__((aligned(16))) __bf16 cH[2][16 * NORD];
  __shared__ __attribute__((aligned(16))) __bf16 cL[2][16 * NORD];
  __shared__ __attribute__((aligned(16))) float  uLds[LCHUNK * 16];

  const int tid    = threadIdx.x;
  const int laneLo = tid & 15;          // column within tile / row within tile
  const int laneHi = (tid >> 4) & 1;    // VGPR-half selector (ISA 7.12.2)
  const int wave   = tid >> 5;          // 16 waves, wave w owns cols 16w..16w+15
  const int n0     = wave * 16;
  const int b0     = blockIdx.y * 16;   // batch tile
  const int chunk  = blockIdx.x;
  const int nCol   = n0 + laneLo;
  const int stOff  = (laneHi * 8) * NORD + nCol;  // state publish base offset

  // Preload this chunk's driving inputs into LDS (rank-1 modes).
  if (MODE != 2) {
    const int t0 = chunk * STEPS;
    __builtin_prefetch(&uSrc[(size_t)t0 * BATCH + b0], 0, 1);
    for (int i = tid; i < STEPS * 16; i += 512) {
      int t = i >> 4, m = i & 15;
      uLds[i] = uSrc[(size_t)(t0 + t) * BATCH + b0 + m];
    }
  }

  // Step-invariant B-operands (dA^T column tile) pinned in VGPRs: 8 k-tiles,
  // hi+lo -> 16 * v16bf = 128 VGPRs per wave.
  v16bf bHf[8], bLf[8];
#pragma unroll
  for (int kt = 0; kt < 8; ++kt) {
    v16bf h, l;
#pragma unroll
    for (int j = 0; j < 16; ++j) {
      int kl = (j < 8) ? (laneHi * 8 + j) : (16 + laneHi * 8 + (j - 8));
      int k  = kt * 32 + kl;
      h[j] = Bh[k * NORD + nCol];
      l[j] = Bl[k * NORD + nCol];
    }
    bHf[kt] = h;
    bLf[kt] = l;
  }

  float dBv = 0.f;
  if (MODE != 2) dBv = dB[nCol];

  // Initial state (previous-step accumulator).
  v8f acc;
  if (MODE == 3 && chunk > 0) {
    const float* ip =
        initS + ((size_t)(chunk - 1) * BATCH + b0 + laneHi * 8) * NORD + nCol;
#pragma unroll
    for (int r = 0; r < 8; ++r) acc[r] = ip[(size_t)r * NORD];
  } else {
#pragma unroll
    for (int r = 0; r < 8; ++r) acc[r] = 0.f;
  }

  // Publish initial state to buffer 0 (bf16 hi/lo split).
#pragma unroll
  for (int r = 0; r < 8; ++r) {
    float  x = acc[r];
    __bf16 h = (__bf16)x;
    cH[0][stOff + r * NORD] = h;
    cL[0][stOff + r * NORD] = (__bf16)(x - (float)h);
  }
  __syncthreads();

#pragma unroll 1
  for (int t = 0; t < STEPS; ++t) {
    const int rd = t & 1;
    const int wr = rd ^ 1;
    const __bf16* rH = cH[rd];
    const __bf16* rL = cL[rd];

    // Three independent accumulator chains: aH*bH, aL*bH, aH*bL.
    v8f a0, a1, a2;
    if (MODE == 2) {
      const float* up =
          uSrc + ((size_t)t * BATCH + b0 + laneHi * 8) * NORD + nCol;
#pragma unroll
      for (int r = 0; r < 8; ++r) a0[r] = up[(size_t)r * NORD];
    } else {
#pragma unroll
      for (int r = 0; r < 8; ++r)
        a0[r] = uLds[t * 16 + laneHi * 8 + r] * dBv;
    }
#pragma unroll
    for (int r = 0; r < 8; ++r) { a1[r] = 0.f; a2[r] = 0.f; }

    // 8 k-tiles x 3 split products = 24 v_wmma_f32_16x16x32_bf16 / step / wave
#pragma unroll
    for (int kt = 0; kt < 8; ++kt) {
      const int abase = laneLo * NORD + kt * 32 + laneHi * 8;
      v16bf aH = load_a_frag(&rH[abase]);
      v16bf aL = load_a_frag(&rL[abase]);
      a0 = __builtin_amdgcn_wmma_f32_16x16x32_bf16(
          false, aH, false, bHf[kt], (short)0, a0, false, false);
      a1 = __builtin_amdgcn_wmma_f32_16x16x32_bf16(
          false, aL, false, bHf[kt], (short)0, a1, false, false);
      a2 = __builtin_amdgcn_wmma_f32_16x16x32_bf16(
          false, aH, false, bLf[kt], (short)0, a2, false, false);
    }

    // Combine chains -> new state; publish to the write buffer.
#pragma unroll
    for (int r = 0; r < 8; ++r) {
      float  x = a0[r] + a1[r] + a2[r];
      __bf16 h = (__bf16)x;
      cH[wr][stOff + r * NORD] = h;
      cL[wr][stOff + r * NORD] = (__bf16)(x - (float)h);
      acc[r] = x;
    }

    if (MODE != 1) {
      size_t tg  = (size_t)chunk * STEPS + t;
      float* op  = outAll + (tg * BATCH + b0 + laneHi * 8) * NORD + nCol;
#pragma unroll
      for (int r = 0; r < 8; ++r) op[(size_t)r * NORD] = acc[r];
    }
    __syncthreads();
  }

  if (MODE == 1) {
    float* op =
        outFinal + ((size_t)chunk * BATCH + b0 + laneHi * 8) * NORD + nCol;
#pragma unroll
    for (int r = 0; r < 8; ++r) op[(size_t)r * NORD] = acc[r];
  }
}

// ---------------------------------------------------------------------------
extern "C" void kernel_launch(void* const* d_in, const int* in_sizes, int n_in,
                              void* d_out, int out_size, void* d_ws,
                              size_t ws_size, hipStream_t stream) {
  (void)in_sizes; (void)n_in; (void)out_size; (void)ws_size;
  const float* inputs = (const float*)d_in[0];  // (L,B)
  const float* dA     = (const float*)d_in[1];  // (N,N)
  const float* dB     = (const float*)d_in[2];  // (N,)
  float* out = (float*)d_out;

  char* ws = (char*)d_ws;
  float*  pA     = (float*)ws;                              // 256 KB
  float*  pB     = (float*)(ws + 256 * 1024);               // 256 KB
  __bf16* ATh    = (__bf16*)(ws + 512 * 1024);              // 128 KB each
  __bf16* ATl    = ATh + NORD * NORD;
  __bf16* ApTh   = ATl + NORD * NORD;
  __bf16* ApTl   = ApTh + NORD * NORD;
  float*  sFinal = (float*)(ws + 1024 * 1024);              // 4 MB (C,B,N)
  float*  Sarr   = sFinal + NCHUNK * BATCH * NORD;          // 4 MB (C,B,N)

  dim3 mb(16, 16), mg(NORD / 16, NORD / 16);
  // dA^128 by repeated squaring: ^2 ^4 ^8 ^16 ^32 ^64 ^128 -> ends in pA
  hippo_sq<<<mg, mb, 0, stream>>>(dA, dA, pA);
  hippo_sq<<<mg, mb, 0, stream>>>(pA, pA, pB);
  hippo_sq<<<mg, mb, 0, stream>>>(pB, pB, pA);
  hippo_sq<<<mg, mb, 0, stream>>>(pA, pA, pB);
  hippo_sq<<<mg, mb, 0, stream>>>(pB, pB, pA);
  hippo_sq<<<mg, mb, 0, stream>>>(pA, pA, pB);
  hippo_sq<<<mg, mb, 0, stream>>>(pB, pB, pA);

  hippo_tsplit<<<NORD, 256, 0, stream>>>(dA, ATh, ATl);
  hippo_tsplit<<<NORD, 256, 0, stream>>>(pA, ApTh, ApTl);

  // Pass 1: chunk-local finals (zero init).
  hippo_scan<1, LCHUNK><<<dim3(NCHUNK, BATCH / 16), 512, 0, stream>>>(
      inputs, dB, ATh, ATl, nullptr, nullptr, sFinal);
  // Pass 2: combine across chunks through dA^128 (a 16-step scan).
  hippo_scan<2, NCHUNK><<<dim3(1, BATCH / 16), 512, 0, stream>>>(
      sFinal, nullptr, ApTh, ApTl, nullptr, Sarr, nullptr);
  // Pass 3: re-scan each chunk seeded with true carry, stream all outputs.
  hippo_scan<3, LCHUNK><<<dim3(NCHUNK, BATCH / 16), 512, 0, stream>>>(
      inputs, dB, ATh, ATl, Sarr, out, nullptr);
}
